// PatternAggregator_72129680769654
// MI455X (gfx1250) — matrix operand order repaired and checked by reading, not compiled, not run on hardware
//
#include <hip/hip_runtime.h>

typedef _Float16 v16h __attribute__((ext_vector_type(16)));
typedef _Float16 v8h  __attribute__((ext_vector_type(8)));
typedef float    v8f  __attribute__((ext_vector_type(8)));
typedef float    v4f  __attribute__((ext_vector_type(4)));

#define SDIM 512
#define DDIM 256
#define HS_STRIDE 264   // halfs: scaled h stage row stride (528B rows: 16B-aligned, bank-step 4)
#define HT_STRIDE 40    // halfs: transposed h stage row stride (80B rows: 16B-aligned, bank-step 20)
#define AL_STRIDE 32    // halfs: alpha stage row stride

__device__ __forceinline__ v16h cat16(v8h lo, v8h hi) {
    return __builtin_shufflevector(lo, hi, 0,1,2,3,4,5,6,7,8,9,10,11,12,13,14,15);
}

__global__ __launch_bounds__(256)
void gat_fused_kernel(const float* __restrict__ hidden,
                      const int*   __restrict__ adj,
                      const float* __restrict__ a0, const float* __restrict__ a1,
                      const float* __restrict__ a2, const float* __restrict__ a3,
                      const float* __restrict__ bias,
                      float* __restrict__ out)
{
    __shared__ alignas(16) _Float16 a_lds[4 * DDIM];            //  2048 B : a_k as f16
    __shared__ alignas(16) _Float16 hs[4][16 * HS_STRIDE];      // 33792 B : a_k-scaled h[J:J+16,:] row-major
    __shared__ alignas(16) _Float16 h_tr[DDIM * HT_STRIDE];     // 20480 B : raw h transposed [d][j0..j31]
    __shared__ alignas(16) _Float16 alpha_st[8][16 * AL_STRIDE];//  8192 B : per-wave 16x32 alpha tiles

    const int tid  = threadIdx.x;
    const int lane = tid & 31;
    const int wave = tid >> 5;
    const int b      = blockIdx.x >> 2;
    const int rowblk = blockIdx.x & 3;
    const int I      = rowblk * 128 + wave * 16;   // this wave's 16-row tile

    const int lm = lane & 15;
    const int lh = lane >> 4;

    const float* hb   = hidden + (size_t)b * SDIM * DDIM;
    const int*   adjb = adj    + (size_t)b * SDIM * SDIM;

    // ---- a_k -> LDS as f16 (256 threads cover D exactly) ----
    a_lds[0*DDIM + tid] = (_Float16)a0[tid];
    a_lds[1*DDIM + tid] = (_Float16)a1[tid];
    a_lds[2*DDIM + tid] = (_Float16)a2[tid];
    a_lds[3*DDIM + tid] = (_Float16)a3[tid];

    // ---- A-operand fragments of raw h_I (16x32 per K-chunk): only 64 VGPRs ----
    // 16-bit A layout: lane (lh,lm): row m=lm, elems 0..7 = K lh*8+{0..7}, elems 8..15 = K 16+lh*8+{0..7}
    v16h hA[8];
    #pragma unroll
    for (int c = 0; c < 8; ++c) {
        const float* hp = hb + (size_t)(I + lm) * DDIM + c * 32 + lh * 8;
        v4f h0 = *(const v4f*)(hp + 0);
        v4f h1 = *(const v4f*)(hp + 4);
        v4f h2 = *(const v4f*)(hp + 16);
        v4f h3 = *(const v4f*)(hp + 20);
        v16h q;
        #pragma unroll
        for (int i = 0; i < 4; ++i) {
            q[i]      = (_Float16)h0[i];
            q[4 + i]  = (_Float16)h1[i];
            q[8 + i]  = (_Float16)h2[i];
            q[12 + i] = (_Float16)h3[i];
        }
        hA[c] = q;
    }

    v8f oacc[16];
    #pragma unroll
    for (int dc = 0; dc < 16; ++dc) oacc[dc] = {};

    // cooperative staging coords: lanes sweep j contiguously (bank-friendly h_tr stores,
    // and lane l / l+16 pair covers one full 128B line of a row)
    const int sj = lane & 15;                 // j within the 16-row chunk
    const int sd = (wave * 2 + lh) * 16;      // 16-wide d slice

    for (int jb = 0; jb < 16; ++jb) {
        #pragma unroll
        for (int p = 0; p < 2; ++p) {
            const int J = jb * 32 + p * 16;

            // adj tile loads: hoisted global loads, latency hidden under staging + WMMA
            int adjr[8];
            #pragma unroll
            for (int r = 0; r < 8; ++r)
                adjr[r] = adjb[(size_t)(I + r + lh * 8) * SDIM + J + lm];

            __syncthreads();   // previous-phase readers done before restaging

            // ---- stage h[b, J:J+16, :] -> h_tr (raw) + 4 a_k-scaled row-major copies ----
            {
                const float* gp = hb + (size_t)(J + sj) * DDIM + sd;
                v4f f0 = *(const v4f*)(gp + 0);
                v4f f1 = *(const v4f*)(gp + 4);
                v4f f2 = *(const v4f*)(gp + 8);
                v4f f3 = *(const v4f*)(gp + 12);
                v8h tl, th;
                #pragma unroll
                for (int i = 0; i < 4; ++i) {
                    tl[i]     = (_Float16)f0[i];
                    tl[4 + i] = (_Float16)f1[i];
                    th[i]     = (_Float16)f2[i];
                    th[4 + i] = (_Float16)f3[i];
                }
                #pragma unroll
                for (int i = 0; i < 8; ++i) {
                    h_tr[(sd + i)     * HT_STRIDE + p * 16 + sj] = tl[i];
                    h_tr[(sd + 8 + i) * HT_STRIDE + p * 16 + sj] = th[i];
                }
                #pragma unroll
                for (int k = 0; k < 4; ++k) {
                    v8h alo = *(const v8h*)(a_lds + k * DDIM + sd);
                    v8h ahi = *(const v8h*)(a_lds + k * DDIM + sd + 8);
                    *(v8h*)(&hs[k][sj * HS_STRIDE + sd])     = tl * alo;  // v_pk_mul_f16
                    *(v8h*)(&hs[k][sj * HS_STRIDE + sd + 8]) = th * ahi;
                }
            }
            __syncthreads();

            // ---- GEMM-1: e_k tile = h_I (A) x (a_k*h_J)^T (B), merged per-k into araw ----
            float araw[8];
            #pragma unroll
            for (int r = 0; r < 8; ++r) araw[r] = 0.0f;
            #pragma unroll
            for (int k = 0; k < 4; ++k) {
                v8f e = {};
                #pragma unroll
                for (int c = 0; c < 8; ++c) {
                    // B (32x16): column n=lm is scaled row lm of h_J; lane K = c*32 + lh*8 + {0..7}, c*32 + 16 + lh*8 + {0..7}
                    const _Float16* bp = &hs[k][lm * HS_STRIDE + c * 32 + lh * 8];
                    v16h Bf = cat16(*(const v8h*)bp, *(const v8h*)(bp + 16));
                    e = __builtin_amdgcn_wmma_f32_16x16x32_f16(false, hA[c], false, Bf, (short)0, e, false, false);
                }
                #pragma unroll
                for (int r = 0; r < 8; ++r)
                    araw[r] = (adjr[r] == k + 1) ? e[r] : araw[r];
            }
            // activation + stage alpha tile column half (C layout: m = r + lh*8, n = lm)
            #pragma unroll
            for (int r = 0; r < 8; ++r) {
                float x  = araw[r];
                float lv = x >= 0.0f ? x : 0.2f * x;
                float sg = __builtin_amdgcn_rcpf(1.0f + __expf(-lv));   // fast sigmoid
                float al = ((unsigned)(adjr[r] - 1) < 4u) ? sg : 0.0f;  // sigmoid(-FLT_MAX)==0
                alpha_st[wave][(r + lh * 8) * AL_STRIDE + p * 16 + lm] = (_Float16)al;
            }
        } // phases

        // ---- GEMM-2: out_tile += alpha(16x32) @ h(32x256) ----
        const _Float16* apb = &alpha_st[wave][lm * AL_STRIDE + lh * 8];
        v16h Af = cat16(*(const v8h*)apb, *(const v8h*)(apb + 16));
        #pragma unroll
        for (int dc = 0; dc < 16; ++dc) {
            const _Float16* bp = h_tr + (dc * 16 + lm) * HT_STRIDE + lh * 8;
            v16h Bf = cat16(*(const v8h*)bp, *(const v8h*)(bp + 16));
            oacc[dc] = __builtin_amdgcn_wmma_f32_16x16x32_f16(false, Af, false, Bf, (short)0, oacc[dc], false, false);
        }
    }

    // ---- epilogue: add bias, store f32 output tile ----
    #pragma unroll
    for (int dc = 0; dc < 16; ++dc) {
        float bz = bias[dc * 16 + lm];
        #pragma unroll
        for (int r = 0; r < 8; ++r)
            out[((size_t)b * SDIM + I + r + lh * 8) * DDIM + dc * 16 + lm] = oacc[dc][r] + bz;
    }
}

extern "C" void kernel_launch(void* const* d_in, const int* in_sizes, int n_in,
                              void* d_out, int out_size, void* d_ws, size_t ws_size,
                              hipStream_t stream) {
    (void)in_sizes; (void)n_in; (void)d_ws; (void)ws_size; (void)out_size;
    const float* hidden = (const float*)d_in[0];
    const int*   adj    = (const int*)  d_in[1];
    const float* a0     = (const float*)d_in[2];
    const float* a1     = (const float*)d_in[3];
    const float* a2     = (const float*)d_in[4];
    const float* a3     = (const float*)d_in[5];
    const float* bias   = (const float*)d_in[6];
    float* out = (float*)d_out;

    // 64 batches x 4 row-blocks (8 waves x 16 rows each) = 256 blocks of 256 threads
    gat_fused_kernel<<<dim3(256), dim3(256), 0, stream>>>(hidden, adj, a0, a1, a2, a3, bias, out);
}
